// _PointnetSAModuleBase_5085241279177
// MI455X (gfx1250) — matrix operand order repaired and checked by reading, not compiled, not run on hardware
//
#include <hip/hip_runtime.h>

typedef __attribute__((ext_vector_type(16))) _Float16 v16h;
typedef __attribute__((ext_vector_type(8)))  float    v8f;

#define NB   4
#define NPTS 16384
#define NP   1024
#define NS   64
#define R2   0.09f          // f32(0.3*0.3 in double) == 0x3DB851EC

union V16H { v16h v; _Float16 h[16]; };

__device__ __forceinline__ float dist2rn(float dx, float dy, float dz) {
  // match XLA: square (round-to-nearest, no FMA contraction), then left-to-right sum
  return __fadd_rn(__fadd_rn(__fmul_rn(dx, dx), __fmul_rn(dy, dy)), __fmul_rn(dz, dz));
}

// ---- WMMA fragment loaders (layouts per CDNA5 ISA 7.12.2, wave32) -------

// A-matrix 16x32 f16 tile, row-major in LDS with row stride ldk.
__device__ __forceinline__ v16h load_a(const _Float16* Wt, int ldk, int lane) {
  int m  = lane & 15;
  int kg = (lane >> 4) & 1;
  V16H a;
#pragma unroll
  for (int i = 0; i < 16; ++i) {
    int k = ((i >> 3) << 4) + (kg << 3) + (i & 7);
    a.h[i] = Wt[m * ldk + k];
  }
  return a.v;
}

// B-matrix 32x16 f16 tile from row-major [K][ld] LDS, columns col..col+15.
__device__ __forceinline__ v16h load_b(const _Float16* X, int ld, int col, int lane) {
  int n  = lane & 15;
  int kb = (lane >> 4) << 3;
  V16H bf;
#pragma unroll
  for (int i = 0; i < 16; ++i) bf.h[i] = X[(((i >> 3) << 4) + kb + (i & 7)) * ld + col + n];
  return bf.v;
}

// ======================= Kernel 1: furthest point sampling ===============
// One block per batch, 256 threads. mindist in LDS (64KB); each thread keeps
// its 64 points' coordinates in registers (only 4 blocks -> occupancy moot).
// Argmax reduction carries the winner's coordinates, so the per-iteration
// critical path has zero global round trips: VALU + LDS + 2 barriers.
__global__ __launch_bounds__(256) void fps_kernel(const float* __restrict__ xyz,
                                                  float* out) {
  const int b    = blockIdx.x;
  const int tid  = threadIdx.x;
  const int lane = tid & 31;
  const int wave = tid >> 5;
  const float* px    = xyz + (size_t)b * NPTS * 3;
  float*       newxy = out + (size_t)b * NP * 3;

  __shared__ float mind[NPTS];                 // 64 KB
  __shared__ float wv[8], wx[8], wy[8], wz[8];
  __shared__ int   wi[8];
  __shared__ float s_lx, s_ly, s_lz;

  // preload this thread's 64 points into registers
  float cx[64], cy[64], cz[64];
#pragma unroll
  for (int k = 0; k < 64; ++k) {
    int j = tid + (k << 8);
    cx[k] = px[j * 3 + 0];
    cy[k] = px[j * 3 + 1];
    cz[k] = px[j * 3 + 2];
  }
#pragma unroll
  for (int k = 0; k < 64; ++k) mind[tid + (k << 8)] = 1.0e10f;

  float lx = px[0], ly = px[1], lz = px[2];    // first sample = point 0
  if (tid == 0) { newxy[0] = lx; newxy[1] = ly; newxy[2] = lz; }
  __syncthreads();

  for (int it = 1; it < NP; ++it) {
    float bestv = -1.0f, bx = 0.f, by = 0.f, bz = 0.f;
    int   besti = 0;
#pragma unroll
    for (int k = 0; k < 64; ++k) {
      int   j  = tid + (k << 8);
      float dx = cx[k] - lx, dy = cy[k] - ly, dz = cz[k] - lz;
      float d  = dist2rn(dx, dy, dz);
      float m  = mind[j];
      m = (d < m) ? d : m;
      mind[j] = m;
      bool better = (m > bestv);               // first-max: j strictly increases
      bestv = better ? m : bestv;
      besti = better ? j : besti;
      bx = better ? cx[k] : bx;
      by = better ? cy[k] : by;
      bz = better ? cz[k] : bz;
    }
    // wave-level argmax butterfly (carry coordinates along)
#pragma unroll
    for (int off = 16; off > 0; off >>= 1) {
      float ov = __shfl_xor(bestv, off, 32);
      int   oi = __shfl_xor(besti, off, 32);
      float ox = __shfl_xor(bx, off, 32);
      float oy = __shfl_xor(by, off, 32);
      float oz = __shfl_xor(bz, off, 32);
      if (ov > bestv || (ov == bestv && oi < besti)) {
        bestv = ov; besti = oi; bx = ox; by = oy; bz = oz;
      }
    }
    if (lane == 0) { wv[wave] = bestv; wi[wave] = besti;
                     wx[wave] = bx; wy[wave] = by; wz[wave] = bz; }
    __syncthreads();
    if (tid == 0) {
      float bw = wv[0]; int bI = wi[0]; int sel = 0;
      for (int w = 1; w < 8; ++w)
        if (wv[w] > bw || (wv[w] == bw && wi[w] < bI)) { bw = wv[w]; bI = wi[w]; sel = w; }
      s_lx = wx[sel]; s_ly = wy[sel]; s_lz = wz[sel];
      newxy[it * 3 + 0] = wx[sel];
      newxy[it * 3 + 1] = wy[sel];
      newxy[it * 3 + 2] = wz[sel];
    }
    __syncthreads();
    lx = s_lx; ly = s_ly; lz = s_lz;
  }
}

// ============ Kernel 2: ball query + group + 3-layer MLP + max ===========
__global__ __launch_bounds__(128) void group_mlp_kernel(
    const float* __restrict__ xyz, const float* __restrict__ feat,
    const float* __restrict__ W1, const float* __restrict__ b1,
    const float* __restrict__ W2, const float* __restrict__ b2,
    const float* __restrict__ W3, const float* __restrict__ b3,
    float* out) {
  const int blk  = blockIdx.x;
  const int b    = blk >> 10;
  const int p    = blk & 1023;
  const int tid  = threadIdx.x;
  const int lane = tid & 31;
  const int wave = tid >> 5;

  __shared__ _Float16 W1ls[64 * 32];    // K padded 6 -> 32 with zeros
  __shared__ _Float16 W2ls[64 * 64];
  __shared__ _Float16 W3ls[128 * 64];
  __shared__ _Float16 Gls[32 * 64];     // [K=32][col=64], rows 6..31 zero
  __shared__ _Float16 H1ls[64 * 64];
  __shared__ _Float16 H2ls[64 * 64];
  __shared__ unsigned maskls[512];
  __shared__ int      idxls[NS];
  __shared__ float    partls[4 * 128];
  __shared__ float    b1ls[64], b2ls[64], b3ls[128];

  // ---- stage weights (f16) and biases ----
  for (int i = tid; i < 64 * 32; i += 128) {
    int m = i >> 5, k = i & 31;
    W1ls[i] = (k < 6) ? (_Float16)W1[m * 6 + k] : (_Float16)0.0f;
  }
  for (int i = tid; i < 64 * 64; i += 128)  W2ls[i] = (_Float16)W2[i];
  for (int i = tid; i < 128 * 64; i += 128) W3ls[i] = (_Float16)W3[i];
  if (tid < 64) { b1ls[tid] = b1[tid]; b2ls[tid] = b2[tid]; }
  if (tid < 128) b3ls[tid] = b3[tid];

  // ---- ball query: build 16384-bit in-ball mask ----
  const float* q  = out + ((size_t)b * NP + p) * 3;   // new_xyz from kernel 1
  const float  qx = q[0], qy = q[1], qz = q[2];
  const float* px = xyz  + (size_t)b * NPTS * 3;
  const float* pf = feat + (size_t)b * NPTS * 3;

  for (int w = wave; w < 512; w += 4) {
    int   j  = w * 32 + lane;
    float dx = px[j * 3 + 0] - qx;
    float dy = px[j * 3 + 1] - qy;
    float dz = px[j * 3 + 2] - qz;
    unsigned long long bal = __ballot(dist2rn(dx, dy, dz) < R2);
    if (lane == 0) maskls[w] = (unsigned)bal;
  }
  __syncthreads();

  // first NS in-ball indices in index order; pad with first (0 if none)
  if (tid == 0) {
    int cnt = 0;
    for (int w = 0; w < 512 && cnt < NS; ++w) {
      unsigned mm = maskls[w];
      while (mm && cnt < NS) {
        int l = __ffs(mm) - 1;
        mm &= mm - 1;
        idxls[cnt++] = w * 32 + l;
      }
    }
    int first = (cnt > 0) ? idxls[0] : 0;
    for (int s = cnt; s < NS; ++s) idxls[s] = first;
  }
  __syncthreads();

  // ---- build grouped input G: rows 0-2 rel xyz, 3-5 features, 6-31 zero ----
  for (int i = 6 * 64 + tid; i < 32 * 64; i += 128) Gls[i] = (_Float16)0.0f;
  if (tid < NS) {
    int j = idxls[tid];
    Gls[0 * 64 + tid] = (_Float16)(px[j * 3 + 0] - qx);
    Gls[1 * 64 + tid] = (_Float16)(px[j * 3 + 1] - qy);
    Gls[2 * 64 + tid] = (_Float16)(px[j * 3 + 2] - qz);
    Gls[3 * 64 + tid] = (_Float16)pf[j * 3 + 0];
    Gls[4 * 64 + tid] = (_Float16)pf[j * 3 + 1];
    Gls[5 * 64 + tid] = (_Float16)pf[j * 3 + 2];
  }
  __syncthreads();

  const int colb = wave * 16;
  const int n    = lane & 15;
  const int mrow = (lane >> 4) << 3;

  // ---- layer 1: 64 x (K=32 padded) x 16 cols ----
  {
    v16h gB = load_b(Gls, 64, colb, lane);
    for (int mt = 0; mt < 4; ++mt) {
      v16h a   = load_a(W1ls + mt * 16 * 32, 32, lane);
      v8f  acc = {};
      acc = __builtin_amdgcn_wmma_f32_16x16x32_f16(false, a, false, gB,
                                                   (short)0, acc, false, false);
      int m0 = mt * 16 + mrow;
#pragma unroll
      for (int r = 0; r < 8; ++r) {
        float v = acc[r] + b1ls[m0 + r];
        H1ls[(m0 + r) * 64 + colb + n] = (_Float16)(v > 0.f ? v : 0.f);
      }
    }
  }
  // ---- layer 2: 64 x 64 x 16 cols ----
  {
    v16h bf0 = load_b(H1ls,           64, colb, lane);
    v16h bf1 = load_b(H1ls + 32 * 64, 64, colb, lane);
    for (int mt = 0; mt < 4; ++mt) {
      v8f acc = {};
      acc = __builtin_amdgcn_wmma_f32_16x16x32_f16(false, load_a(W2ls + mt * 16 * 64,      64, lane),
                                                   false, bf0, (short)0, acc, false, false);
      acc = __builtin_amdgcn_wmma_f32_16x16x32_f16(false, load_a(W2ls + mt * 16 * 64 + 32, 64, lane),
                                                   false, bf1, (short)0, acc, false, false);
      int m0 = mt * 16 + mrow;
#pragma unroll
      for (int r = 0; r < 8; ++r) {
        float v = acc[r] + b2ls[m0 + r];
        H2ls[(m0 + r) * 64 + colb + n] = (_Float16)(v > 0.f ? v : 0.f);
      }
    }
  }
  // ---- layer 3: 128 x 64 x 16 cols, fused relu + max over this wave's cols ----
  {
    v16h bf0 = load_b(H2ls,           64, colb, lane);
    v16h bf1 = load_b(H2ls + 32 * 64, 64, colb, lane);
    for (int mt = 0; mt < 8; ++mt) {
      v8f acc = {};
      acc = __builtin_amdgcn_wmma_f32_16x16x32_f16(false, load_a(W3ls + mt * 16 * 64,      64, lane),
                                                   false, bf0, (short)0, acc, false, false);
      acc = __builtin_amdgcn_wmma_f32_16x16x32_f16(false, load_a(W3ls + mt * 16 * 64 + 32, 64, lane),
                                                   false, bf1, (short)0, acc, false, false);
      int m0 = mt * 16 + mrow;
#pragma unroll
      for (int r = 0; r < 8; ++r) {
        float v = acc[r] + b3ls[m0 + r];
        v = (v > 0.f) ? v : 0.f;
        v = fmaxf(v, __shfl_xor(v, 1, 32));   // max over this wave's 16 columns
        v = fmaxf(v, __shfl_xor(v, 2, 32));
        v = fmaxf(v, __shfl_xor(v, 4, 32));
        v = fmaxf(v, __shfl_xor(v, 8, 32));
        if (n == 0) partls[wave * 128 + m0 + r] = v;
      }
    }
  }
  __syncthreads();

  // ---- max across the 4 waves, write new_features[b, c, p] ----
  {
    float v = partls[tid];
    v = fmaxf(v, partls[128 + tid]);
    v = fmaxf(v, partls[256 + tid]);
    v = fmaxf(v, partls[384 + tid]);
    out[(size_t)NB * NP * 3 + ((size_t)b * 128 + tid) * NP + p] = v;
  }
}

extern "C" void kernel_launch(void* const* d_in, const int* in_sizes, int n_in,
                              void* d_out, int out_size, void* d_ws, size_t ws_size,
                              hipStream_t stream) {
  (void)in_sizes; (void)n_in; (void)out_size; (void)d_ws; (void)ws_size;
  const float* xyz  = (const float*)d_in[0];
  const float* feat = (const float*)d_in[1];
  const float* W1   = (const float*)d_in[2];
  const float* b1   = (const float*)d_in[3];
  const float* W2   = (const float*)d_in[4];
  const float* b2   = (const float*)d_in[5];
  const float* W3   = (const float*)d_in[6];
  const float* b3   = (const float*)d_in[7];
  float* out = (float*)d_out;

  hipLaunchKernelGGL(fps_kernel, dim3(NB), dim3(256), 0, stream, xyz, out);
  hipLaunchKernelGGL(group_mlp_kernel, dim3(NB * NP), dim3(128), 0, stream,
                     xyz, feat, W1, b1, W2, b2, W3, b3, out);
}
